// DLRM_Net_84155589198706
// MI455X (gfx1250) — matrix operand order repaired and checked by reading, not compiled, not run on hardware
//
#include <hip/hip_runtime.h>

// ---------------- problem constants ----------------
#define NT     26       // embedding tables
#define ROWS_T 100001   // rows per table
#define DDIM   64       // embedding / bottom-MLP out dim
#define BATCH  8192
#define LPOOL  4        // bag size
#define NFEAT  27       // 1 dense + 26 sparse features
#define FEATPAD 32      // padded feature count (rows 27..31 zeroed)
#define TFS    (FEATPAD * DDIM)  // 2048 bf16 per sample in T_feat
#define RLD    416      // padded interaction row (64 + 351 -> 416; col 415 is a dump slot)

typedef __bf16 bf16_t;
typedef __attribute__((ext_vector_type(16))) __bf16 v16bf;
typedef __attribute__((ext_vector_type(8)))  __bf16 v8bf;
typedef __attribute__((ext_vector_type(4)))  __bf16 v4bf;
typedef __attribute__((ext_vector_type(8)))  float  v8f;
typedef __attribute__((ext_vector_type(4)))  float  v4f;

static __device__ __forceinline__ bf16_t f2bf(float f) { return (bf16_t)f; }

struct v8bf_pair { v8bf lo, hi; };
static __device__ __forceinline__ v16bf combine(v8bf lo, v8bf hi) {
  v8bf_pair p{lo, hi};
  return __builtin_bit_cast(v16bf, p);   // pure register concat
}

// ---------------------------------------------------------------------------
// WMMA operand layouts used by this file (wave32; contiguous k order, same for A and B):
// A 16x32 bf16: element e of lane -> M = lane%16 (+tile),
//               K = k0 + 16*(lane/16) + e
//   => per lane: one contiguous group of 16 K-values (two 16B halves)
// B 32x16 bf16: element e of lane -> N = lane%16 (+tile), K = k0 + 16*(lane/16) + e
//   => per lane: one contiguous group of 16 K-values (two 16B halves)
// C/D 16x16 f32: vgpr r, lane -> N = lane%16, M = r + 8*(lane/16)
// ---------------------------------------------------------------------------
template <int LDA>
static __device__ __forceinline__ v16bf load_a_bf(const bf16_t* __restrict__ A,
                                                  int m, int k0, int lane) {
  const int half = lane >> 4;
  const bf16_t* p = A + (size_t)m * LDA + k0 + (half << 4);
  return combine(*(const v8bf*)(p), *(const v8bf*)(p + 8));
}

template <int LDW, int NACT>
static __device__ __forceinline__ v16bf load_b_bf(const bf16_t* __restrict__ W,
                                                  int nBase, int k0, int lane) {
  const int half = lane >> 4;
  int n = nBase + (lane & 15);
  bool nok = true;
  if constexpr ((NACT & 15) != 0) {   // only the N=1 final layer
    nok = (n < NACT);
    n = nok ? n : (NACT - 1);
  }
  const bf16_t* p = W + (size_t)n * LDW + k0 + (half << 4);
  v16bf b = combine(*(const v8bf*)(p), *(const v8bf*)(p + 8));
  if constexpr ((NACT & 15) != 0) {
#pragma unroll
    for (int e = 0; e < 16; ++e) b[e] = nok ? b[e] : f2bf(0.0f);
  }
  return b;
}

#define WMMA_BF16(A_, B_, C_) \
  __builtin_amdgcn_wmma_f32_16x16x32_bf16(false, (A_), false, (B_), (short)0, (C_), false, false)

// ---------------------------------------------------------------------------
// GEMM: C[8192 x NACT] = act(A[8192 x KACT] * W[NACT x KACT]^T + bias).
// bf16 in, f32 accumulate, OutT out. KACT is a multiple of 32 (pre-padded):
// no guards in the k-loop. If NACT % 32 == 0 each wave computes a 64x32 C
// tile (8 wmma / 12 b128 loads per k-step, A reused across two B tiles);
// otherwise (N=1 layer) a 64x16 tile with a branchless n-select.
// ---------------------------------------------------------------------------
template <int KACT, int NACT, int LDA, int LDW, bool RELU, typename OutT>
__global__ void gemm_bias_act_wmma(const bf16_t* __restrict__ A,
                                   const bf16_t* __restrict__ W,
                                   const float* __restrict__ bias,
                                   OutT* __restrict__ C, int ldc) {
  static_assert(KACT % 32 == 0, "K must be pre-padded to a multiple of 32");
  constexpr bool PAIR = ((NACT & 31) == 0);
  constexpr int NTILES = PAIR ? (NACT / 32) : ((NACT + 15) / 16);
  constexpr int MTILES = BATCH / 64;
  constexpr int KFULL = KACT / 32;

  const int lane = threadIdx.x & 31;
  const int wave = threadIdx.x >> 5;
  const int tile = blockIdx.x * 8 + wave;
  if (tile >= MTILES * NTILES) return;  // wave-uniform
  const int tm = tile / NTILES;
  const int tn = tile - tm * NTILES;
  const int mBase = tm << 6;
  const int nBase = PAIR ? (tn << 5) : (tn << 4);
  const int mRow = mBase + (lane & 15);

  v8f acc0 = {}, acc1 = {}, acc2 = {}, acc3 = {};
  v8f acc4 = {}, acc5 = {}, acc6 = {}, acc7 = {};

  for (int ks = 0; ks < KFULL; ++ks) {
    const int k0 = ks << 5;
    v16bf bt0 = load_b_bf<LDW, NACT>(W, nBase, k0, lane);
    v16bf a0 = load_a_bf<LDA>(A, mRow +  0, k0, lane);
    v16bf a1 = load_a_bf<LDA>(A, mRow + 16, k0, lane);
    v16bf a2 = load_a_bf<LDA>(A, mRow + 32, k0, lane);
    v16bf a3 = load_a_bf<LDA>(A, mRow + 48, k0, lane);
    acc0 = WMMA_BF16(a0, bt0, acc0);
    acc1 = WMMA_BF16(a1, bt0, acc1);
    acc2 = WMMA_BF16(a2, bt0, acc2);
    acc3 = WMMA_BF16(a3, bt0, acc3);
    if constexpr (PAIR) {
      v16bf bt1 = load_b_bf<LDW, NACT>(W, nBase + 16, k0, lane);
      acc4 = WMMA_BF16(a0, bt1, acc4);
      acc5 = WMMA_BF16(a1, bt1, acc5);
      acc6 = WMMA_BF16(a2, bt1, acc6);
      acc7 = WMMA_BF16(a3, bt1, acc7);
    }
  }

  const int half = lane >> 4;
  const int n0 = nBase + (lane & 15);
  bool nOK = true;
  int nc = n0;
  if constexpr ((NACT & 15) != 0) {
    nOK = (n0 < NACT);
    nc = nOK ? n0 : (NACT - 1);
  }
  const float bv0 = bias[nc];

#define STORE_SUBTILE(ACC, MOFF, NCOL, BV)                              \
  {                                                                     \
    _Pragma("unroll") for (int r = 0; r < 8; ++r) {                     \
      const int m = mBase + (MOFF) + r + (half << 3);                   \
      if (nOK) {                                                        \
        float v = (ACC)[r] + (BV);                                      \
        if constexpr (RELU) v = fmaxf(v, 0.0f);                         \
        C[(size_t)m * ldc + (NCOL)] = (OutT)v;                          \
      }                                                                 \
    }                                                                   \
  }
  STORE_SUBTILE(acc0, 0, n0, bv0)
  STORE_SUBTILE(acc1, 16, n0, bv0)
  STORE_SUBTILE(acc2, 32, n0, bv0)
  STORE_SUBTILE(acc3, 48, n0, bv0)
  if constexpr (PAIR) {
    const int n1 = n0 + 16;
    const float bv1 = bias[n1];
    STORE_SUBTILE(acc4, 0, n1, bv1)
    STORE_SUBTILE(acc5, 16, n1, bv1)
    STORE_SUBTILE(acc6, 32, n1, bv1)
    STORE_SUBTILE(acc7, 48, n1, bv1)
  }
#undef STORE_SUBTILE
}

// ---------------------------------------------------------------------------
// One-shot f32 -> bf16 conversion with K zero-padding (weights, dense_x).
// ---------------------------------------------------------------------------
__global__ void cvt_pad_bf16(const float* __restrict__ src, bf16_t* __restrict__ dst,
                             int ksrc, int kdst) {
  const int r = blockIdx.y;
  const int k = blockIdx.x * blockDim.x + threadIdx.x;
  if (k >= kdst) return;
  const int kc = (k < ksrc) ? k : (ksrc - 1);
  float v = src[(size_t)r * ksrc + kc];
  v = (k < ksrc) ? v : 0.0f;
  dst[(size_t)r * kdst + k] = f2bf(v);
}

// ---------------------------------------------------------------------------
// Embedding gather + mean (dominant ~218 MB f32 gather traffic).
// One thread per (table, bag, 4-dim group): 4x b128 gathers, bf16 b64 store.
// ---------------------------------------------------------------------------
__global__ void embedding_mean_kernel(const int* __restrict__ lS_i,
                                      const float* __restrict__ tables,
                                      bf16_t* __restrict__ T_feat) {
  const int id = blockIdx.x * blockDim.x + threadIdx.x;
  if (id >= NT * BATCH * 16) return;
  const int dg = id & 15;            // 4-dim group
  const int rest = id >> 4;
  const int b = rest & (BATCH - 1);  // BATCH = 2^13
  const int t = rest >> 13;

  const int* ip = lS_i + (((size_t)t * BATCH + b) << 2);
  const int i0 = ip[0], i1 = ip[1], i2 = ip[2], i3 = ip[3];

  const float* tab = tables + (size_t)t * ROWS_T * DDIM + (dg << 2);
  v4f s = *(const v4f*)(tab + (size_t)i0 * DDIM);
  s = s + *(const v4f*)(tab + (size_t)i1 * DDIM);
  s = s + *(const v4f*)(tab + (size_t)i2 * DDIM);
  s = s + *(const v4f*)(tab + (size_t)i3 * DDIM);
  s = s * 0.25f;
  v4bf o;
#pragma unroll
  for (int i = 0; i < 4; ++i) o[i] = f2bf(s[i]);
  *(v4bf*)(T_feat + (size_t)b * TFS + ((1 + t) << 6) + (dg << 2)) = o;
}

// Zero the padding feature rows 27..31 of every sample (one v8bf per thread).
__global__ void zero_tail_feats(bf16_t* __restrict__ T_feat) {
  const int id = blockIdx.x * blockDim.x + threadIdx.x;
  if (id >= BATCH * (FEATPAD - NFEAT) * 8) return;
  const int c = id & 7;
  const int rest = id >> 3;
  const int s = rest % (FEATPAD - NFEAT);
  const int b = rest / (FEATPAD - NFEAT);
  v8bf z = {};
  *(v8bf*)(T_feat + (size_t)b * TFS + ((NFEAT + s) << 6) + (c << 3)) = z;
}

// ---------------------------------------------------------------------------
// Interaction: per sample, Gram = Tf(32x64 bf16, rows 27..31 zero) * Tf^T.
// Tiles (0,0),(1,0),(1,1) cover all strict-lower m>n pairs.
// Scatter is branchless and uses compile-time triangular offsets: for each
// accumulator register the two possible rows (half=0/1) are constants, so
// both candidate offsets fold; invalid entries dump into R col 415, which
// multiplies only against the zero-padded column of the bf16 top_w0 copy
// (finite * 0 == 0), so its contents are irrelevant.
// ---------------------------------------------------------------------------
static __device__ __forceinline__ v16bf feat_a_tile(const bf16_t* __restrict__ Tf,
                                                    int mBase, int k0, int lane) {
  const int half = lane >> 4;
  const int m = mBase + (lane & 15);
  const bf16_t* p = Tf + (m << 6) + k0 + (half << 4);
  return combine(*(const v8bf*)(p), *(const v8bf*)(p + 8));
}

static __device__ __forceinline__ v16bf feat_b_tile(const bf16_t* __restrict__ Tf,
                                                    int nBase, int k0, int lane) {
  const int half = lane >> 4;
  const int n = nBase + (lane & 15);
  const bf16_t* p = Tf + (n << 6) + k0 + (half << 4);
  return combine(*(const v8bf*)(p), *(const v8bf*)(p + 8));
}

template <int MBASE, int NBASE>
static __device__ __forceinline__ void scatter_tri(const v8f& acc, int lane,
                                                   bf16_t* __restrict__ Rb) {
  const int n = NBASE + (lane & 15);
  const bool hi = (lane & 16) != 0;
#pragma unroll
  for (int r = 0; r < 8; ++r) {
    const int m0 = MBASE + r;        // row when half==0 (constant after unroll)
    const int m1 = MBASE + r + 8;    // row when half==1 (constant after unroll)
    const int tri0 = DDIM + (m0 * (m0 - 1)) / 2;   // folds to an immediate
    const int tri1 = DDIM + (m1 * (m1 - 1)) / 2;   // folds to an immediate
    const bool ok0 = (m0 < NFEAT) && (n < m0);     // m0<NFEAT folds
    const bool ok1 = (m1 < NFEAT) && (n < m1);     // m1<NFEAT folds
    const int off0 = ok0 ? (tri0 + n) : (RLD - 1);
    const int off1 = ok1 ? (tri1 + n) : (RLD - 1);
    Rb[hi ? off1 : off0] = f2bf(acc[r]);
  }
}

__global__ void interaction_wmma(const bf16_t* __restrict__ T_feat,
                                 bf16_t* __restrict__ R) {
  const int lane = threadIdx.x & 31;
  const int wave = threadIdx.x >> 5;
  const int b = blockIdx.x * 8 + wave;
  if (b >= BATCH) return;
  const bf16_t* Tf = T_feat + (size_t)b * TFS;

  v8f acc00 = {}, acc10 = {}, acc11 = {};
#pragma unroll
  for (int ks = 0; ks < 2; ++ks) {
    const int k0 = ks << 5;
    v16bf a0 = feat_a_tile(Tf, 0, k0, lane);
    v16bf a1 = feat_a_tile(Tf, 16, k0, lane);
    v16bf b0 = feat_b_tile(Tf, 0, k0, lane);
    v16bf b1 = feat_b_tile(Tf, 16, k0, lane);
    acc00 = WMMA_BF16(a0, b0, acc00);
    acc10 = WMMA_BF16(a1, b0, acc10);
    acc11 = WMMA_BF16(a1, b1, acc11);
  }

  bf16_t* Rb = R + (size_t)b * RLD;
  // R[:,0:64] = x : 64 bf16 = 32 dwords, one dword per lane, no loop
  ((unsigned*)Rb)[lane] = ((const unsigned*)Tf)[lane];

  scatter_tri<0, 0>(acc00, lane, Rb);
  scatter_tri<16, 0>(acc10, lane, Rb);
  scatter_tri<16, 16>(acc11, lane, Rb);
}

// ---------------------------------------------------------------------------
// host-side launch
// ---------------------------------------------------------------------------
template <int KACT, int NACT, int LDA, int LDW, bool RELU, typename OutT>
static inline void launch_gemm(const bf16_t* A, const bf16_t* W, const float* bias,
                               OutT* C, int ldc, hipStream_t stream) {
  constexpr int NTILES = ((NACT & 31) == 0) ? (NACT / 32) : ((NACT + 15) / 16);
  constexpr int tiles = (BATCH / 64) * NTILES;
  const int blocks = (tiles + 7) / 8;  // 8 waves / block (256 threads)
  gemm_bias_act_wmma<KACT, NACT, LDA, LDW, RELU, OutT>
      <<<blocks, 256, 0, stream>>>(A, W, bias, C, ldc);
}

static inline void launch_cvt(const float* src, bf16_t* dst, int rows, int ksrc, int kdst,
                              hipStream_t stream) {
  dim3 grid((kdst + 255) / 256, rows);
  cvt_pad_bf16<<<grid, 256, 0, stream>>>(src, dst, ksrc, kdst);
}

extern "C" void kernel_launch(void* const* d_in, const int* in_sizes, int n_in,
                              void* d_out, int out_size, void* d_ws, size_t ws_size,
                              hipStream_t stream) {
  const float* dense_x = (const float*)d_in[0];
  // d_in[1] = lS_o (unused: offsets are b*L by construction)
  const int*   lS_i    = (const int*)d_in[2];
  const float* emb     = (const float*)d_in[3];
  const float* bot_w0  = (const float*)d_in[4];
  const float* bot_b0  = (const float*)d_in[5];
  const float* bot_w1  = (const float*)d_in[6];
  const float* bot_b1  = (const float*)d_in[7];
  const float* bot_w2  = (const float*)d_in[8];
  const float* bot_b2  = (const float*)d_in[9];
  const float* top_w0  = (const float*)d_in[10];
  const float* top_b0  = (const float*)d_in[11];
  const float* top_w1  = (const float*)d_in[12];
  const float* top_b1  = (const float*)d_in[13];
  const float* top_w2  = (const float*)d_in[14];
  const float* top_b2  = (const float*)d_in[15];
  float* out = (float*)d_out;

  // bf16 workspace carve-up (all region sizes are multiples of 64 elements)
  bf16_t* p = (bf16_t*)d_ws;
  bf16_t* xb   = p; p += (size_t)BATCH * 32;    // dense_x padded 13->32
  bf16_t* wb0  = p; p += (size_t)512 * 32;      // bot_w0 padded
  bf16_t* wb1  = p; p += (size_t)256 * 512;
  bf16_t* wb2  = p; p += (size_t)64 * 256;
  bf16_t* wt0  = p; p += (size_t)512 * RLD;     // top_w0 padded 415->416 (col 415 zero)
  bf16_t* wt1  = p; p += (size_t)256 * 512;
  bf16_t* wt2  = p; p += (size_t)1 * 256;
  bf16_t* x0b  = p; p += (size_t)BATCH * 512;   // bottom act 0 (aliases z0)
  bf16_t* x1b  = p; p += (size_t)BATCH * 256;   // bottom act 1 (aliases z1)
  bf16_t* Tfb  = p; p += (size_t)BATCH * TFS;   // 8192 x 32 x 64
  bf16_t* Rb   = p; p += (size_t)BATCH * RLD;   // 8192 x 416
  bf16_t* z0b = x0b;
  bf16_t* z1b = x1b;
  (void)ws_size; (void)in_sizes; (void)n_in; (void)out_size;

  // ---- one-shot bf16 staging (weights + dense input, K padded to 32x) ----
  launch_cvt(dense_x, xb, BATCH, 13, 32, stream);
  launch_cvt(bot_w0, wb0, 512, 13, 32, stream);
  launch_cvt(bot_w1, wb1, 256, 512, 512, stream);
  launch_cvt(bot_w2, wb2, 64, 256, 256, stream);
  launch_cvt(top_w0, wt0, 512, 415, RLD, stream);
  launch_cvt(top_w1, wt1, 256, 512, 512, stream);
  launch_cvt(top_w2, wt2, 1, 256, 256, stream);

  // ---- bottom MLP (bf16 -> bf16) ----
  launch_gemm<32, 512, 32, 32, true>(xb, wb0, bot_b0, x0b, 512, stream);
  launch_gemm<512, 256, 512, 512, true>(x0b, wb1, bot_b1, x1b, 256, stream);
  // layer 2 writes straight into T_feat feature slot 0 (ldc = TFS)
  launch_gemm<256, 64, 256, 256, true>(x1b, wb2, bot_b2, Tfb, TFS, stream);

  // ---- embedding gather + mean -> T_feat slots 1..26; zero slots 27..31 ----
  {
    const int total = NT * BATCH * 16;
    embedding_mean_kernel<<<(total + 255) / 256, 256, 0, stream>>>(lS_i, emb, Tfb);
    const int ztotal = BATCH * (FEATPAD - NFEAT) * 8;
    zero_tail_feats<<<(ztotal + 255) / 256, 256, 0, stream>>>(Tfb);
  }

  // ---- pairwise interaction -> R (8192 x 416 bf16) ----
  interaction_wmma<<<BATCH / 8, 256, 0, stream>>>(Tfb, Rb);

  // ---- top MLP (final layer writes f32 to d_out) ----
  launch_gemm<RLD, 512, RLD, RLD, true>(Rb, wt0, top_b0, z0b, 512, stream);
  launch_gemm<512, 256, 512, 512, true>(z0b, wt1, top_b1, z1b, 256, stream);
  launch_gemm<256, 1, 256, 256, false>(z1b, wt2, top_b2, out, 1, stream);
}
